// PICInputNet_71622874628531
// MI455X (gfx1250) — matrix operand order repaired and checked by reading, not compiled, not run on hardware
//
#include <hip/hip_runtime.h>
#include <hip/hip_bf16.h>

// ---------------------------------------------------------------------------
// PICInputNet fused kernel for MI455X (gfx1250, wave32, WMMA).
//
//   feat = [cos, sin](2*pi * z ⊗ coeff)        (1024 x 64)   -> f16 in d_ws
//   h_v  = tanh(feat @ w1_v^T)                 (1024 x 64)   -> per-wave LDS
//   out_v = h_v @ w2_v^T                       (1024 x 128)  -> f32 global
//
// One block per variable v (784 blocks, 256 threads = 8 waves).
// Each wave owns 8 Q-tiles of 16 rows for BOTH gemms -> fully fused, the
// 205 MB intermediate h never touches HBM. Output stores (411 MB) dominate.
// ---------------------------------------------------------------------------

typedef __attribute__((ext_vector_type(16))) _Float16 v16h;
typedef __attribute__((ext_vector_type(8)))  _Float16 v8h;
typedef __attribute__((ext_vector_type(8)))  float    v8f;

#define NQ 1024
#define NV 784
#define NP 128
#define ND 64
#define NFF 64

__device__ __forceinline__ float fast_tanh(float x) {
  // CDNA5 hardware transcendental (TRANS unit, co-executes with WMMA)
  float r;
  asm volatile("v_tanh_f32 %0, %1" : "=v"(r) : "v"(x));
  return r;
}

// ---- Kernel 1: Fourier features, f32 -> f16, 1024x64 -----------------------
__global__ void feat_kernel(const float* __restrict__ z,
                            const float* __restrict__ coeff,
                            _Float16* __restrict__ feat) {
  int idx = blockIdx.x * blockDim.x + threadIdx.x;   // Q * FF/2 = 32768
  if (idx >= NQ * (NFF / 2)) return;
  int q = idx >> 5;
  int f = idx & 31;
  float zp = 6.283185307179586f * z[q] * coeff[f];
  float s, c;
  sincosf(zp, &s, &c);
  feat[q * NFF + f]      = (_Float16)c;
  feat[q * NFF + 32 + f] = (_Float16)s;
}

// ---- Kernel 2: fused per-variable dual GEMM --------------------------------
__global__ void __launch_bounds__(256)
picnet_kernel(const _Float16* __restrict__ feat,
              const float* __restrict__ w1,
              const float* __restrict__ w2,
              float* __restrict__ out) {
  __shared__ _Float16 ldsW1[ND * NFF];     // [d][k]  8 KB  (B-operand gemm1)
  __shared__ _Float16 ldsW2[NP * ND];      // [p][d] 16 KB  (B-operand gemm2)
  __shared__ _Float16 ldsH[8 * 16 * ND];   // per-wave 16x64 h tile, 16 KB

  const int v    = blockIdx.x;
  const int tid  = threadIdx.x;
  const int lane = tid & 31;
  const int wave = tid >> 5;
  const int nlo  = lane & 15;   // row (A) / col (B,D) within 16
  const int hi   = lane >> 4;   // lane-group select per WMMA VGPR layout

  // Stage this variable's weights into LDS as f16 (natural row-major layout
  // is exactly the WMMA B-operand friendly layout: two contiguous 8-K chunks per lane).
  const float* w1g = w1 + (size_t)v * (ND * NFF);
  for (int i = tid; i < ND * NFF; i += 256) ldsW1[i] = (_Float16)w1g[i];
  const float* w2g = w2 + (size_t)v * (NP * ND);
  for (int i = tid; i < NP * ND; i += 256) ldsW2[i] = (_Float16)w2g[i];
  __syncthreads();

  _Float16* hrow = ldsH + wave * (16 * ND);
  float* outv = out + (size_t)v * (NQ * NP);

  for (int qt = wave; qt < NQ / 16; qt += 8) {
    const int q0 = qt * 16;

    // ---------------- GEMM1: h(16x64) = feat(16x64) @ W1^T, tanh ----------
    v8f acc1[4] = {};
#pragma unroll
    for (int ks = 0; ks < 2; ++ks) {
      const int k0 = ks * 32;
      // A fragment (16x32 f16): two contiguous 16B chunks per lane
      const _Float16* arow = feat + (size_t)(q0 + nlo) * NFF + k0 + hi * 8;
      v8h a_lo = *(const v8h*)(arow);
      v8h a_hi = *(const v8h*)(arow + 16);
      v16h a = __builtin_shufflevector(a_lo, a_hi,
                0,1,2,3,4,5,6,7,8,9,10,11,12,13,14,15);
#pragma unroll
      for (int t = 0; t < 4; ++t) {       // N-tiles over D = 4 x 16
        const _Float16* brow = ldsW1 + (t * 16 + nlo) * NFF + k0 + hi * 8;
        v8h b0 = *(const v8h*)(brow);
        v8h b1 = *(const v8h*)(brow + 16);
        v16h b = __builtin_shufflevector(b0, b1,
                  0,1,2,3,4,5,6,7,8,9,10,11,12,13,14,15);
        acc1[t] = __builtin_amdgcn_wmma_f32_16x16x32_f16(
            false, a, false, b, (short)0, acc1[t], false, false);
      }
    }

    // tanh (HW trans op) + cvt f16, park h tile row-major in this wave's LDS
#pragma unroll
    for (int t = 0; t < 4; ++t) {
#pragma unroll
      for (int r = 0; r < 8; ++r) {
        hrow[(r + 8 * hi) * ND + t * 16 + nlo] = (_Float16)fast_tanh(acc1[t][r]);
      }
    }
    // same-wave LDS RAW: make stores visible before ds_load of A2 fragments
    asm volatile("s_wait_dscnt 0x0" ::: "memory");

    // ---------------- GEMM2: out(16x128) = h(16x64) @ W2^T ----------------
    v8f acc2[8] = {};
#pragma unroll
    for (int ks = 0; ks < 2; ++ks) {
      const int k0 = ks * 32;
      const _Float16* arow = hrow + nlo * ND + k0 + hi * 8;
      v8h a_lo = *(const v8h*)(arow);
      v8h a_hi = *(const v8h*)(arow + 16);
      v16h a = __builtin_shufflevector(a_lo, a_hi,
                0,1,2,3,4,5,6,7,8,9,10,11,12,13,14,15);
#pragma unroll
      for (int t = 0; t < 8; ++t) {       // N-tiles over P = 8 x 16
        const _Float16* brow = ldsW2 + (t * 16 + nlo) * ND + k0 + hi * 8;
        v8h b0 = *(const v8h*)(brow);
        v8h b1 = *(const v8h*)(brow + 16);
        v16h b = __builtin_shufflevector(b0, b1,
                  0,1,2,3,4,5,6,7,8,9,10,11,12,13,14,15);
        acc2[t] = __builtin_amdgcn_wmma_f32_16x16x32_f16(
            false, a, false, b, (short)0, acc2[t], false, false);
      }
    }

    // Store out[v, q0+m, :]; each half-wave writes 64B contiguous segments
#pragma unroll
    for (int r = 0; r < 8; ++r) {
      float* orow = outv + (size_t)(q0 + r + 8 * hi) * NP + nlo;
#pragma unroll
      for (int t = 0; t < 8; ++t) {
        orow[t * 16] = acc2[t][r];
      }
    }
  }
}

extern "C" void kernel_launch(void* const* d_in, const int* in_sizes, int n_in,
                              void* d_out, int out_size, void* d_ws, size_t ws_size,
                              hipStream_t stream) {
  const float* z     = (const float*)d_in[0];   // (1024,)
  const float* coeff = (const float*)d_in[1];   // (1, 32)
  const float* w1    = (const float*)d_in[2];   // (D*V, FF)
  const float* w2    = (const float*)d_in[3];   // (P*V, D)
  float* out = (float*)d_out;                   // (V, Q, P) f32
  _Float16* feat = (_Float16*)d_ws;             // 1024*64 f16 = 128 KB scratch

  feat_kernel<<<(NQ * (NFF / 2) + 255) / 256, 256, 0, stream>>>(z, coeff, feat);
  picnet_kernel<<<NV, 256, 0, stream>>>(feat, w1, w2, out);
}
